// CacheAttention_69913477644571
// MI455X (gfx1250) — compile-verified, matrix operand order checked by reading, not run on hardware
//
#include <hip/hip_runtime.h>
#include <hip/hip_bf16.h>

// ---------------------------------------------------------------------------
// CacheAttention for MI455X (gfx1250, wave32, WMMA).
//   hs[4096,4096] @ Wq^T -> q (scaled)          : bf16 WMMA GEMM
//   retr[65536,1024] @ Wk^T / Wv^T -> k, v      : bf16 WMMA GEMM
//   per (b,s,kv): softmax(q k^T) v  (tiny)      : fp32 VALU
//   ctx[4096,4096] @ Wo^T -> out (fp32)         : bf16 WMMA GEMM
// Compute-bound (~1000 FLOP/byte): bf16 16x16x32 WMMA w/ f32 accumulate.
// Staging uses gfx1250 async global->LDS copies (ASYNCcnt) with LDS double
// buffering so global latency overlaps WMMA issue.
// ---------------------------------------------------------------------------

typedef __attribute__((ext_vector_type(16))) __bf16 bf16x16;
typedef __attribute__((ext_vector_type(8)))  __bf16 bf16x8;
typedef __attribute__((ext_vector_type(4)))  __bf16 bf16x4;
typedef __attribute__((ext_vector_type(8)))  float  f32x8;
typedef __attribute__((ext_vector_type(4)))  int    i32x4;

#define LDST 40   // padded LDS row stride in bf16 elems (80 B: 16B-aligned, bank-spread)

#if defined(__gfx1250__) &&                                              \
    __has_builtin(__builtin_amdgcn_global_load_async_to_lds_b128) &&     \
    __has_builtin(__builtin_amdgcn_s_wait_asynccnt)
#define USE_ASYNC_LDS 1
#else
#define USE_ASYNC_LDS 0
#endif

#if USE_ASYNC_LDS
typedef __attribute__((address_space(1))) i32x4* gas_i32x4_ptr;
typedef __attribute__((address_space(3))) i32x4* lds_i32x4_ptr;

__device__ __forceinline__ void async_b128(const __bf16* g, __bf16* l) {
  // AS(1) = global (flat addr == global addr);
  // AS(3) = LDS: low 32 bits of the flat shared address are the wave-relative
  //         LDS byte offset, which is what GLOBAL_LOAD_ASYNC_TO_LDS's VDST wants.
  __builtin_amdgcn_global_load_async_to_lds_b128(
      (gas_i32x4_ptr)(uintptr_t)g,
      (lds_i32x4_ptr)(unsigned)(uintptr_t)l,
      0, 0);
}
#endif

// ---------------- fp32 -> bf16 streaming cast (one pass) -------------------
__global__ void cast_f32_to_bf16(const float* __restrict__ in,
                                 __bf16* __restrict__ out, int n4) {
  int i = blockIdx.x * blockDim.x + threadIdx.x;
  if (i < n4) {
    float4 v = ((const float4*)in)[i];
    bf16x4 o;
    o[0] = (__bf16)v.x; o[1] = (__bf16)v.y; o[2] = (__bf16)v.z; o[3] = (__bf16)v.w;
    ((bf16x4*)out)[i] = o;
  }
}

// ---------------- bf16 WMMA GEMM: Out[M,N] = A[M,K] * W[N,K]^T * scale -----
// Block: 256 thr (8 waves). Block tile 128x128, BK=32, LDS double-buffered.
// Wave tile 64x32 (4x2 wmma accumulators).
template <bool OUT_BF16>
__global__ __launch_bounds__(256) void gemm_bf16_wmma(
    const __bf16* __restrict__ A,   // [M,K] row-major
    const __bf16* __restrict__ W,   // [N,K] row-major (nn.Linear weight)
    void* __restrict__ Out,         // [M,N] bf16 or f32
    int M, int N, int K, float scale) {
  __shared__ __align__(16) __bf16 As[2][128 * LDST];
  __shared__ __align__(16) __bf16 Bs[2][128 * LDST];

  const int tid  = threadIdx.x;
  const int lane = tid & 31;
  const int wave = tid >> 5;
  const int wm   = wave & 1;        // 64-row strip within block tile
  const int wn   = wave >> 1;       // 32-col strip within block tile
  const int half = lane >> 4;       // ISA 16-bit A/B layout: lane half picks K chunks
  const int lr   = lane & 15;

  const int mblock = blockIdx.y * 128;
  const int nblock = blockIdx.x * 128;

  f32x8 acc[4][2];
#pragma unroll
  for (int mi = 0; mi < 4; ++mi)
#pragma unroll
    for (int ni = 0; ni < 2; ++ni)
#pragma unroll
      for (int r = 0; r < 8; ++r) acc[mi][ni][r] = 0.0f;

  // Staging: 128 rows x 32 bf16 = 512 x 16B chunks per matrix; 2 per thread.
  const int r0 = tid >> 2;          // 0..63
  const int c0 = tid & 3;           // 16B chunk in row

  // Issue one block-tile of copies (4 per thread: 2 A + 2 B).
  auto stage = [&](int kt, int buf) {
    const __bf16* Ag = A + (size_t)mblock * K + (size_t)kt * 32;
    const __bf16* Wg = W + (size_t)nblock * K + (size_t)kt * 32;
#if USE_ASYNC_LDS
#pragma unroll
    for (int i = 0; i < 2; ++i) {
      const int r = r0 + 64 * i;
      async_b128(Ag + (size_t)r * K + c0 * 8, &As[buf][r * LDST + c0 * 8]);
      async_b128(Wg + (size_t)r * K + c0 * 8, &Bs[buf][r * LDST + c0 * 8]);
    }
#else
#pragma unroll
    for (int i = 0; i < 2; ++i) {
      const int r = r0 + 64 * i;
      bf16x8 a = *(const bf16x8*)(Ag + (size_t)r * K + c0 * 8);
      bf16x8 b = *(const bf16x8*)(Wg + (size_t)r * K + c0 * 8);
      *(bf16x8*)(&As[buf][r * LDST + c0 * 8]) = a;
      *(bf16x8*)(&Bs[buf][r * LDST + c0 * 8]) = b;
    }
#endif
  };

  const int nkt = K >> 5;
  stage(0, 0);

  for (int kt = 0; kt < nkt; ++kt) {
    const int cur = kt & 1;
#if USE_ASYNC_LDS
    if (kt + 1 < nkt) {
      stage(kt + 1, cur ^ 1);                 // prefetch next tile (other buffer)
      __builtin_amdgcn_s_wait_asynccnt(4);    // retire this tile's 4 copies
    } else {
      __builtin_amdgcn_s_wait_asynccnt(0);
    }
#endif
    __syncthreads();                          // tile `cur` visible to all waves

    // Fragment loads per CDNA5 16-bit A/B VGPR layout:
    //  elements 0..7  -> k = half*8 + 0..7
    //  elements 8..15 -> k = 16 + half*8 + 0..7
    bf16x16 afrag[4], bfrag[2];
#pragma unroll
    for (int mi = 0; mi < 4; ++mi) {
      const __bf16* p = &As[cur][(wm * 64 + mi * 16 + lr) * LDST + half * 8];
      bf16x8 lo = *(const bf16x8*)p;
      bf16x8 hi = *(const bf16x8*)(p + 16);
#pragma unroll
      for (int e = 0; e < 8; ++e) { afrag[mi][e] = lo[e]; afrag[mi][e + 8] = hi[e]; }
    }
#pragma unroll
    for (int ni = 0; ni < 2; ++ni) {
      const __bf16* p = &Bs[cur][(wn * 32 + ni * 16 + lr) * LDST + half * 8];
      bf16x8 lo = *(const bf16x8*)p;
      bf16x8 hi = *(const bf16x8*)(p + 16);
#pragma unroll
      for (int e = 0; e < 8; ++e) { bfrag[ni][e] = lo[e]; bfrag[ni][e + 8] = hi[e]; }
    }

#pragma unroll
    for (int mi = 0; mi < 4; ++mi)
#pragma unroll
      for (int ni = 0; ni < 2; ++ni)
        acc[mi][ni] = __builtin_amdgcn_wmma_f32_16x16x32_bf16(
            false, afrag[mi], false, bfrag[ni], (short)0, acc[mi][ni], false, false);

    __syncthreads();                          // LDS reads done before buffer reuse
#if !USE_ASYNC_LDS
    if (kt + 1 < nkt) stage(kt + 1, cur ^ 1); // sync fallback: stage into other buffer
#endif
  }

  // Epilogue: D vgpr r -> (m = r + 8*half, n = lr) within 16x16 tile.
#pragma unroll
  for (int mi = 0; mi < 4; ++mi)
#pragma unroll
    for (int ni = 0; ni < 2; ++ni)
#pragma unroll
      for (int r = 0; r < 8; ++r) {
        const int row = mblock + wm * 64 + mi * 16 + r + 8 * half;
        const int col = nblock + wn * 32 + ni * 16 + lr;
        const float v = acc[mi][ni][r] * scale;
        if (OUT_BF16) ((__bf16*)Out)[(size_t)row * N + col] = (__bf16)v;
        else          ((float*)Out)[(size_t)row * N + col]  = v;
      }
}

// ---------------- grouped attention core (per (b,s,kv) = one wave) --------
// Replicates torch flat-view: k[b,s,kv,nb,d] = Kmat[bs*16 + 2*kv + nb/8][(nb%8)*128 + d]
__global__ __launch_bounds__(256) void cache_attn_core(
    const __bf16* __restrict__ Q,    // [BS, 4096], head layout (kv*4+g)*128+d, pre-scaled
    const __bf16* __restrict__ Kt,   // [BS*16, 1024]
    const __bf16* __restrict__ Vt,   // [BS*16, 1024]
    __bf16* __restrict__ Ctx) {      // [BS, 4096]
  const int m    = blockIdx.x;        // b*S + s
  const int lane = threadIdx.x & 31;  // wave32: 4 d-elems per lane
  const int kv   = threadIdx.x >> 5;  // 8 waves = 8 kv heads

  float qr[4][4];
  const __bf16* qp = Q + (size_t)m * 4096 + kv * 512;
#pragma unroll
  for (int g = 0; g < 4; ++g) {
    bf16x4 v = *(const bf16x4*)(qp + g * 128 + lane * 4);
#pragma unroll
    for (int j = 0; j < 4; ++j) qr[g][j] = (float)v[j];
  }

  float sc[4][16];
#pragma unroll
  for (int nb = 0; nb < 16; ++nb) {
    const __bf16* kp = Kt + ((size_t)m * 16 + 2 * kv + (nb >> 3)) * 1024 +
                       (nb & 7) * 128 + lane * 4;
    bf16x4 kw = *(const bf16x4*)kp;
    float kf[4];
#pragma unroll
    for (int j = 0; j < 4; ++j) kf[j] = (float)kw[j];
#pragma unroll
    for (int g = 0; g < 4; ++g) {
      float p = qr[g][0] * kf[0] + qr[g][1] * kf[1] + qr[g][2] * kf[2] + qr[g][3] * kf[3];
#pragma unroll
      for (int off = 16; off >= 1; off >>= 1) p += __shfl_xor(p, off, 32);
      sc[g][nb] = p;   // full butterfly: all lanes hold the dot product
    }
  }

#pragma unroll
  for (int g = 0; g < 4; ++g) {
    float mx = sc[g][0];
#pragma unroll
    for (int nb = 1; nb < 16; ++nb) mx = fmaxf(mx, sc[g][nb]);
    float sum = 0.0f;
#pragma unroll
    for (int nb = 0; nb < 16; ++nb) { sc[g][nb] = __expf(sc[g][nb] - mx); sum += sc[g][nb]; }
    const float inv = 1.0f / sum;
#pragma unroll
    for (int nb = 0; nb < 16; ++nb) sc[g][nb] *= inv;
  }

  float acc[4][4];
#pragma unroll
  for (int g = 0; g < 4; ++g)
#pragma unroll
    for (int j = 0; j < 4; ++j) acc[g][j] = 0.0f;

#pragma unroll
  for (int nb = 0; nb < 16; ++nb) {
    const __bf16* vp = Vt + ((size_t)m * 16 + 2 * kv + (nb >> 3)) * 1024 +
                       (nb & 7) * 128 + lane * 4;
    bf16x4 vw = *(const bf16x4*)vp;
    float vf[4];
#pragma unroll
    for (int j = 0; j < 4; ++j) vf[j] = (float)vw[j];
#pragma unroll
    for (int g = 0; g < 4; ++g)
#pragma unroll
      for (int j = 0; j < 4; ++j) acc[g][j] = fmaf(sc[g][nb], vf[j], acc[g][j]);
  }

  __bf16* op = Ctx + (size_t)m * 4096 + kv * 512;
#pragma unroll
  for (int g = 0; g < 4; ++g) {
    bf16x4 o;
#pragma unroll
    for (int j = 0; j < 4; ++j) o[j] = (__bf16)acc[g][j];
    *(bf16x4*)(op + g * 128 + lane * 4) = o;
  }
}

// ---------------------------------------------------------------------------
extern "C" void kernel_launch(void* const* d_in, const int* in_sizes, int n_in,
                              void* d_out, int out_size, void* d_ws, size_t ws_size,
                              hipStream_t stream) {
  const float* hs   = (const float*)d_in[0];  // [2,2048,4096]
  const float* retr = (const float*)d_in[1];  // [2,2048,16,1024]
  const float* Wq   = (const float*)d_in[2];  // [4096,4096]
  const float* Wk   = (const float*)d_in[3];  // [1024,1024]
  const float* Wv   = (const float*)d_in[4];  // [1024,1024]
  const float* Wo   = (const float*)d_in[5];  // [4096,4096]

  constexpr size_t BS  = 2 * 2048;   // 4096
  constexpr size_t HID = 4096;
  constexpr size_t RH  = 1024;
  constexpr size_t NN  = 16;
  constexpr size_t KVD = 1024;       // KV*D
  constexpr size_t HD  = 4096;       // H*D

  char* ws = (char*)d_ws;
  size_t off = 0;
  auto carve = [&](size_t bytes) {
    void* p = ws + off;
    off += (bytes + 255) & ~(size_t)255;
    return p;
  };
  __bf16* hs_b   = (__bf16*)carve(BS * HID * 2);          // 32 MB
  __bf16* retr_b = (__bf16*)carve(BS * NN * RH * 2);      // 128 MB
  __bf16* wq_b   = (__bf16*)carve(HD * HID * 2);          // 32 MB
  __bf16* wk_b   = (__bf16*)carve(KVD * RH * 2);          // 2 MB
  __bf16* wv_b   = (__bf16*)carve(KVD * RH * 2);          // 2 MB
  __bf16* wo_b   = (__bf16*)carve(HID * HD * 2);          // 32 MB
  __bf16* q_b    = (__bf16*)carve(BS * HD * 2);           // 32 MB
  __bf16* k_b    = (__bf16*)carve(BS * NN * KVD * 2);     // 128 MB
  __bf16* v_b    = (__bf16*)carve(BS * NN * KVD * 2);     // 128 MB
  __bf16* ctx_b  = (__bf16*)carve(BS * HD * 2);           // 32 MB

  auto cast = [&](const float* src, __bf16* dst, size_t n) {
    int n4 = (int)(n / 4);
    cast_f32_to_bf16<<<(n4 + 255) / 256, 256, 0, stream>>>(src, dst, n4);
  };
  cast(hs,   hs_b,   BS * HID);
  cast(retr, retr_b, BS * NN * RH);
  cast(Wq,   wq_b,   HD * HID);
  cast(Wk,   wk_b,   KVD * RH);
  cast(Wv,   wv_b,   KVD * RH);
  cast(Wo,   wo_b,   HID * HD);

  const float scaler = 0.08838834764831845f;  // 128^-0.5

  // q = (hs @ Wq^T) * scaler : [4096,4096]
  gemm_bf16_wmma<true><<<dim3(HD / 128, BS / 128), 256, 0, stream>>>(
      hs_b, wq_b, q_b, (int)BS, (int)HD, (int)HID, scaler);
  // k = retr @ Wk^T : [65536,1024]
  gemm_bf16_wmma<true><<<dim3(KVD / 128, (BS * NN) / 128), 256, 0, stream>>>(
      retr_b, wk_b, k_b, (int)(BS * NN), (int)KVD, (int)RH, 1.0f);
  // v = retr @ Wv^T : [65536,1024]
  gemm_bf16_wmma<true><<<dim3(KVD / 128, (BS * NN) / 128), 256, 0, stream>>>(
      retr_b, wv_b, v_b, (int)(BS * NN), (int)KVD, (int)RH, 1.0f);

  // attention core: one block per (b,s), one wave per kv head
  cache_attn_core<<<(int)BS, 256, 0, stream>>>(q_b, k_b, v_b, ctx_b);

  // out = ctx @ Wo^T : [4096,4096] fp32 -> d_out
  gemm_bf16_wmma<false><<<dim3(HID / 128, BS / 128), 256, 0, stream>>>(
      ctx_b, wo_b, d_out, (int)BS, (int)HID, (int)HD, 1.0f);
}
